// HymbaAttention_88828513616496
// MI455X (gfx1250) — compile-verified, matrix operand order checked by reading, not run on hardware
//
#include <hip/hip_runtime.h>
#include <hip/hip_bf16.h>
#include <math.h>

typedef __attribute__((ext_vector_type(16))) _Float16 v16h;
typedef __attribute__((ext_vector_type(8)))  _Float16 v8h;
typedef __attribute__((ext_vector_type(8)))  float    v8f;

#define DIM   1024
#define NH    16
#define KVH   4
#define HD    64
#define II    2048
#define NS    16
#define DTR   64
#define SEQ   2048
#define BB    2
#define MROWS 4096   // B*S

// ---------------------------------------------------------------------------
// Fragment loader: 16-bit A/B WMMA fragment for a row-major row.
// Lane (m = lane&15, khalf = lane>>4). Halves i=0..7 -> K = khalf*8 + i,
// i=8..15 -> K = 16 + khalf*8 + (i-8).  p must point at row + khalf*8,
// 16-byte aligned.
// ---------------------------------------------------------------------------
__device__ __forceinline__ v16h load_frag_row(const _Float16* p) {
  v8h lo = *reinterpret_cast<const v8h*>(p);
  v8h hi = *reinterpret_cast<const v8h*>(p + 16);
  v16h r;
#pragma unroll
  for (int i = 0; i < 8; ++i) { r[i] = lo[i]; r[i + 8] = hi[i]; }
  return r;
}

// ---------------------------------------------------------------------------
// Elementwise f32 -> f16
// ---------------------------------------------------------------------------
__global__ void cvt_f16(const float* __restrict__ in, _Float16* __restrict__ out, size_t n) {
  size_t i = (size_t)blockIdx.x * blockDim.x + threadIdx.x;
  if (i < n) out[i] = (_Float16)in[i];
}

// ---------------------------------------------------------------------------
// Generic GEMM: C[M,N] (f32) = A[M,K] (f16, lda) @ W[N,K]^T (f16, ldw).
// One wave computes a 64x32 tile: 4 M-subtiles x 2 N-subtiles, so each
// 32-K step does 8 WMMAs on 6 fragment loads (A frags reused across the two
// B frags).  Grid: (M/64)*(N/32)/8 blocks of 256 threads (8 waves).
// ---------------------------------------------------------------------------
__global__ void gemm_wmma(const _Float16* __restrict__ A, int lda,
                          const _Float16* __restrict__ W, int ldw,
                          float* __restrict__ C, int M, int N, int K) {
  const int lane = threadIdx.x & 31;
  const int wave = threadIdx.x >> 5;
  const int tile = blockIdx.x * 8 + wave;
  const int ntn  = N >> 5;
  const int m0   = (tile / ntn) << 6;
  const int n0   = (tile % ntn) << 5;
  const int mrow = lane & 15, khalf = lane >> 4;

  v8f acc[4][2] = {};
  const _Float16* wrow0 = W + (size_t)(n0 + mrow) * ldw + khalf * 8;
  const _Float16* wrow1 = wrow0 + (size_t)16 * ldw;
  const _Float16* arow0 = A + (size_t)(m0 + mrow) * lda + khalf * 8;

  for (int k0 = 0; k0 < K; k0 += 32) {
    v16h b0 = load_frag_row(wrow0 + k0);
    v16h b1 = load_frag_row(wrow1 + k0);
#pragma unroll
    for (int t = 0; t < 4; ++t) {
      v16h a = load_frag_row(arow0 + (size_t)t * 16 * lda + k0);
      acc[t][0] = __builtin_amdgcn_wmma_f32_16x16x32_f16(false, a, false, b0,
                                                         (short)0, acc[t][0], false, false);
      acc[t][1] = __builtin_amdgcn_wmma_f32_16x16x32_f16(false, a, false, b1,
                                                         (short)0, acc[t][1], false, false);
    }
  }
  const int cm  = khalf << 3;
  const int cn0 = n0 + mrow;
#pragma unroll
  for (int t = 0; t < 4; ++t)
#pragma unroll
    for (int j = 0; j < 2; ++j)
#pragma unroll
      for (int r = 0; r < 8; ++r)
        C[(size_t)(m0 + t * 16 + cm + r) * N + cn0 + j * 16] = acc[t][j][r];
}

// ---------------------------------------------------------------------------
// Q/K prep: per-head RMSNorm + RoPE (+ q_gain), write f16 [B,H,S,HD] /
// [B,KVH,S,HD]. One wave per (b,s,head); lane d handles pair (d, d+32).
// ---------------------------------------------------------------------------
__global__ void qkv_prep(const float* __restrict__ q_lin, const float* __restrict__ kv_lin,
                         const float* __restrict__ q_gain,
                         _Float16* __restrict__ qh, _Float16* __restrict__ kh) {
  const int lane = threadIdx.x & 31;
  const int wave = threadIdx.x >> 5;
  const int job  = blockIdx.x * 8 + wave;
  const int head = job % (NH + KVH);
  const int bs   = job / (NH + KVH);
  const int b = bs >> 11, s = bs & (SEQ - 1);

  const float* src;
  _Float16*    dst;
  float gain = 1.0f;
  if (head < NH) {
    src  = q_lin + (size_t)bs * DIM + head * HD;
    dst  = qh + (((size_t)b * NH + head) * SEQ + s) * HD;
    gain = q_gain[head];
  } else {
    const int kh_ = head - NH;
    src = kv_lin + (size_t)bs * (2 * KVH * HD) + kh_ * HD;
    dst = kh + (((size_t)b * KVH + kh_) * SEQ + s) * HD;
  }
  float x1 = src[lane], x2 = src[lane + 32];
  float ss = x1 * x1 + x2 * x2;
#pragma unroll
  for (int m = 16; m >= 1; m >>= 1) ss += __shfl_xor(ss, m, 32);
  const float rn = rsqrtf(ss * (1.0f / 64.0f) + 1.1920929e-07f);
  x1 *= rn; x2 *= rn;
  // inv = 10000^{-lane/32}
  const float inv = __expf(-(float)lane * (9.210340371976184f / 32.0f));
  const float f = (float)s * inv;
  const float c = __cosf(f), sn = __sinf(f);
  dst[lane]      = (_Float16)((x1 * c + x2 * sn) * gain);
  dst[lane + 32] = (_Float16)((-x1 * sn + x2 * c) * gain);
}

// V prep: transpose to [B,KVH,HD,S] f16 so PV B-fragments load contiguously.
__global__ void v_prep(const float* __restrict__ kv_lin, _Float16* __restrict__ vt) {
  size_t idx = (size_t)blockIdx.x * 256 + threadIdx.x;   // B*KVH*HD*S
  const int s = idx & (SEQ - 1);
  size_t r = idx >> 11;
  const int d = r & (HD - 1); r >>= 6;
  const int kvh = r & (KVH - 1);
  const int b = (int)(r >> 2);
  vt[idx] = (_Float16)kv_lin[(((size_t)b * SEQ + s) * (2 * KVH * HD)) + KVH * HD + kvh * HD + d];
}

// ---------------------------------------------------------------------------
// Causal flash attention, one wave per 16-query tile, 32-key blocks.
// QK^T and P*V via v_wmma_f32_16x16x32_f16. P transposed C->A layout via LDS.
// ---------------------------------------------------------------------------
__global__ void flash_attn(const _Float16* __restrict__ Q, const _Float16* __restrict__ Kh,
                           const _Float16* __restrict__ Vt, float* __restrict__ out) {
  __shared__ __align__(16) _Float16 pshare[8][16 * 32];
  const int lane = threadIdx.x & 31;
  const int wave = threadIdx.x >> 5;
  const int job  = blockIdx.x * 8 + wave;
  const int qt = job & (SEQ / 16 - 1);
  const int h  = (job >> 7) & (NH - 1);
  const int b  = job >> 11;
  const int q0 = qt << 4;
  const int kvh = h / (NH / KVH);

  const _Float16* qbase = Q  + (((size_t)b * NH  + h)   * SEQ + q0) * HD;
  const _Float16* kbase = Kh + (((size_t)b * KVH + kvh) * SEQ) * HD;
  const _Float16* vbase = Vt + (((size_t)b * KVH + kvh) * HD) * SEQ;
  _Float16* pb = &pshare[wave][0];

  const int mrow = lane & 15, khalf = lane >> 4;
  const v16h qa0 = load_frag_row(qbase + mrow * HD + khalf * 8);
  const v16h qa1 = load_frag_row(qbase + mrow * HD + 32 + khalf * 8);

  v8f o[4] = {};
  float mrun[8], lrun[8];
#pragma unroll
  for (int r = 0; r < 8; ++r) { mrun[r] = -INFINITY; lrun[r] = 0.0f; }
  const float scale = 0.125f;  // 1/sqrt(64)

  for (int kb = 0; kb < q0 + 16; kb += 32) {
    v8f s0 = {}, s1 = {};
    {
      v16h k0a = load_frag_row(kbase + (size_t)(kb + mrow) * HD + khalf * 8);
      v16h k0b = load_frag_row(kbase + (size_t)(kb + mrow) * HD + 32 + khalf * 8);
      s0 = __builtin_amdgcn_wmma_f32_16x16x32_f16(false, qa0, false, k0a, (short)0, s0, false, false);
      s0 = __builtin_amdgcn_wmma_f32_16x16x32_f16(false, qa1, false, k0b, (short)0, s0, false, false);
      v16h k1a = load_frag_row(kbase + (size_t)(kb + 16 + mrow) * HD + khalf * 8);
      v16h k1b = load_frag_row(kbase + (size_t)(kb + 16 + mrow) * HD + 32 + khalf * 8);
      s1 = __builtin_amdgcn_wmma_f32_16x16x32_f16(false, qa0, false, k1a, (short)0, s1, false, false);
      s1 = __builtin_amdgcn_wmma_f32_16x16x32_f16(false, qa1, false, k1b, (short)0, s1, false, false);
    }
    const int kk = kb + (lane & 15);
    const int qb = q0 + ((lane >> 4) << 3);
    float a8[8];
#pragma unroll
    for (int r = 0; r < 8; ++r) {
      const int qq = qb + r;
      float v0 = (kk      > qq) ? -INFINITY : s0[r] * scale;
      float v1 = (kk + 16 > qq) ? -INFINITY : s1[r] * scale;
      float rm = fmaxf(v0, v1);
#pragma unroll
      for (int m = 8; m >= 1; m >>= 1) rm = fmaxf(rm, __shfl_xor(rm, m, 32));
      const float mnew = fmaxf(mrun[r], rm);
      const float p0 = __expf(v0 - mnew);
      const float p1 = __expf(v1 - mnew);
      float ps = p0 + p1;
#pragma unroll
      for (int m = 8; m >= 1; m >>= 1) ps += __shfl_xor(ps, m, 32);
      const float al = __expf(mrun[r] - mnew);
      lrun[r] = lrun[r] * al + ps;
      mrun[r] = mnew;
      a8[r] = al;
      const int row = r + ((lane >> 4) << 3);
      pb[row * 32 + (lane & 15)]      = (_Float16)p0;
      pb[row * 32 + 16 + (lane & 15)] = (_Float16)p1;
    }
#pragma unroll
    for (int t = 0; t < 4; ++t)
#pragma unroll
      for (int r = 0; r < 8; ++r) o[t][r] *= a8[r];

    asm volatile("s_wait_dscnt 0" ::: "memory");  // same-wave LDS RAW
    const v16h pa = load_frag_row(pb + mrow * 32 + khalf * 8);
#pragma unroll
    for (int t = 0; t < 4; ++t) {
      v16h vb = load_frag_row(vbase + (size_t)(t * 16 + mrow) * SEQ + kb + khalf * 8);
      o[t] = __builtin_amdgcn_wmma_f32_16x16x32_f16(false, pa, false, vb, (short)0, o[t], false, false);
    }
  }
#pragma unroll
  for (int t = 0; t < 4; ++t)
#pragma unroll
    for (int r = 0; r < 8; ++r) {
      const int row = r + ((lane >> 4) << 3);
      out[((size_t)b * SEQ + q0 + row) * DIM + h * HD + t * 16 + (lane & 15)] = o[t][r] / lrun[r];
    }
}

// ---------------------------------------------------------------------------
// Depthwise causal conv (K=4) + SiLU.  u f32 + f16 copy for the next GEMM.
// Grid (B*S, I/256), 256 threads.
// ---------------------------------------------------------------------------
__global__ void conv_silu(const float* __restrict__ mp, const float* __restrict__ cw,
                          const float* __restrict__ cb,
                          float* __restrict__ u, _Float16* __restrict__ uh) {
  const int bs = blockIdx.x;
  const int b = bs >> 11, s = bs & (SEQ - 1);
  const int i = blockIdx.y * 256 + threadIdx.x;
  float acc = cb[i];
#pragma unroll
  for (int t = 0; t < 4; ++t) {
    const int sp = s - 3 + t;
    if (sp >= 0) acc += mp[((size_t)(b * SEQ + sp)) * (2 * II) + i] * cw[i * 4 + t];
  }
  const float r = acc / (1.0f + __expf(-acc));
  u[(size_t)bs * II + i]  = r;
  uh[(size_t)bs * II + i] = (_Float16)r;
}

// delta = softplus(delta_raw + b_dt), in place.
__global__ void softplus_bias(float* __restrict__ d, const float* __restrict__ b_dt, size_t n) {
  size_t idx = (size_t)blockIdx.x * 256 + threadIdx.x;
  if (idx >= n) return;
  const float v = d[idx] + b_dt[idx & (II - 1)];
  d[idx] = (v > 20.0f) ? v : log1pf(__expf(v));
}

// ---------------------------------------------------------------------------
// Selective scan: thread per (b,i) channel holds 16-state.  Per timestep the
// 32 B/C params (128B) are staged into LDS by 8 lanes using the CDNA5 async
// LDS-load path (ASYNCcnt), drained with s_wait_asynccnt before the block
// barrier publishes them.  Fused +u*D and gate SiLU; f16 output for the next
// WMMA GEMM.  Grid (I/256, B), 256 threads.
// ---------------------------------------------------------------------------
__global__ void ssm_scan(const float* __restrict__ delta, const float* __restrict__ params,
                         const float* __restrict__ u, const float* __restrict__ A_log,
                         const float* __restrict__ Dp, const float* __restrict__ mp,
                         _Float16* __restrict__ yh) {
  __shared__ __align__(16) float BC[32];   // [0:16)=Bm, [16:32)=Cm
  const int b = blockIdx.y;
  const int i = blockIdx.x * 256 + threadIdx.x;
  float A[NS], st[NS];
#pragma unroll
  for (int n = 0; n < NS; ++n) { A[n] = -__expf(A_log[(size_t)i * NS + n]); st[n] = 0.0f; }
  const float Di = Dp[i];
  const unsigned ldsbase = (unsigned)(uintptr_t)&BC[0];  // LDS aperture: low 32 bits = LDS offset

  for (int s = 0; s < SEQ; ++s) {
    const size_t bs = (size_t)b * SEQ + s;
    if (threadIdx.x < 8) {
      const unsigned ldsoff = ldsbase + threadIdx.x * 16;
      const void* gsrc = (const void*)(params + bs * (DTR + 2 * NS) + DTR + threadIdx.x * 4);
      asm volatile("global_load_async_to_lds_b128 %0, %1, off"
                   :: "v"(ldsoff), "v"(gsrc) : "memory");
    }
    asm volatile("s_wait_asynccnt 0" ::: "memory");
    __syncthreads();
    const float dt = delta[bs * II + i];
    const float uu = u[bs * II + i];
    float y = 0.0f;
#pragma unroll
    for (int n = 0; n < NS; ++n) {
      st[n] = st[n] * __expf(dt * A[n]) + dt * BC[n] * uu;
      y += st[n] * BC[16 + n];
    }
    y += uu * Di;
    const float g = mp[bs * (2 * II) + II + i];
    y *= g / (1.0f + __expf(-g));
    yh[bs * II + i] = (_Float16)y;
    __syncthreads();
  }
}

// merged = sigmoid(alpha)*attn + (1-sigmoid(alpha))*mamba  -> f16
__global__ void merge_kernel(const float* __restrict__ attn, const float* __restrict__ mamba,
                             const float* __restrict__ alpha_p, _Float16* __restrict__ out,
                             size_t n) {
  size_t idx = (size_t)blockIdx.x * 256 + threadIdx.x;
  if (idx >= n) return;
  const float al = 1.0f / (1.0f + __expf(-alpha_p[0]));
  out[idx] = (_Float16)(al * attn[idx] + (1.0f - al) * mamba[idx]);
}

// ---------------------------------------------------------------------------
extern "C" void kernel_launch(void* const* d_in, const int* in_sizes, int n_in,
                              void* d_out, int out_size, void* d_ws, size_t ws_size,
                              hipStream_t stream) {
  const float* x        = (const float*)d_in[0];
  const float* W_mamba  = (const float*)d_in[1];
  const float* W_q      = (const float*)d_in[2];
  const float* W_kv     = (const float*)d_in[3];
  const float* q_gain   = (const float*)d_in[4];
  const float* conv_w   = (const float*)d_in[5];
  const float* conv_b   = (const float*)d_in[6];
  const float* W_xproj  = (const float*)d_in[7];
  const float* W_dt     = (const float*)d_in[8];
  const float* b_dt     = (const float*)d_in[9];
  const float* A_log    = (const float*)d_in[10];
  const float* Dvec     = (const float*)d_in[11];
  const float* W_mo     = (const float*)d_in[12];
  const float* W_proj   = (const float*)d_in[13];
  const float* m_alpha  = (const float*)d_in[14];

  char* ws = (char*)d_ws;
  size_t off = 0;
  auto alloc = [&](size_t bytes) -> void* {
    void* p = ws + off;
    off = (off + bytes + 255) & ~(size_t)255;
    return p;
  };

  _Float16* xh    = (_Float16*)alloc((size_t)MROWS * DIM * 2);
  _Float16* Wm_h  = (_Float16*)alloc((size_t)(2 * II) * DIM * 2);
  _Float16* Wq_h  = (_Float16*)alloc((size_t)DIM * DIM * 2);
  _Float16* Wkv_h = (_Float16*)alloc((size_t)(2 * KVH * HD) * DIM * 2);
  _Float16* Wxp_h = (_Float16*)alloc((size_t)(DTR + 2 * NS) * II * 2);
  _Float16* Wdt_h = (_Float16*)alloc((size_t)II * DTR * 2);
  _Float16* Wmo_h = (_Float16*)alloc((size_t)DIM * II * 2);
  _Float16* Wp_h  = (_Float16*)alloc((size_t)DIM * DIM * 2);
  float*    mp    = (float*)alloc((size_t)MROWS * 2 * II * 4);
  float*    q_lin = (float*)alloc((size_t)MROWS * DIM * 4);          // reused: merged_h
  float*    kv_ln = (float*)alloc((size_t)MROWS * 2 * KVH * HD * 4); // reused: params_h
  _Float16* qh    = (_Float16*)alloc((size_t)BB * NH * SEQ * HD * 2);
  _Float16* kh    = (_Float16*)alloc((size_t)BB * KVH * SEQ * HD * 2);
  _Float16* vt    = (_Float16*)alloc((size_t)BB * KVH * HD * SEQ * 2);
  float*    attn  = (float*)alloc((size_t)MROWS * DIM * 4);
  float*    u     = (float*)alloc((size_t)MROWS * II * 4);
  _Float16* uh    = (_Float16*)alloc((size_t)MROWS * II * 2);
  float*    prms  = (float*)alloc((size_t)MROWS * (DTR + 2 * NS) * 4);
  float*    delta = (float*)alloc((size_t)MROWS * II * 4);
  _Float16* yh    = (_Float16*)alloc((size_t)MROWS * II * 2);
  float*    mamba = (float*)alloc((size_t)MROWS * DIM * 4);
  _Float16* prmsh = (_Float16*)kv_ln;   // alias (kv dead by then)
  _Float16* mrgd  = (_Float16*)q_lin;   // alias (q_lin dead by then)

  auto cvt = [&](const float* src, _Float16* dst, size_t n) {
    cvt_f16<<<(unsigned)((n + 255) / 256), 256, 0, stream>>>(src, dst, n);
  };
  auto gemm = [&](const _Float16* A, int lda, const _Float16* W, int ldw,
                  float* C, int M, int N, int K) {
    const unsigned blocks = (unsigned)((size_t)(M / 64) * (N / 32) / 8);
    gemm_wmma<<<blocks, 256, 0, stream>>>(A, lda, W, ldw, C, M, N, K);
  };

  // 1. f16 conversions
  cvt(x, xh, (size_t)MROWS * DIM);
  cvt(W_mamba, Wm_h, (size_t)(2 * II) * DIM);
  cvt(W_q, Wq_h, (size_t)DIM * DIM);
  cvt(W_kv, Wkv_h, (size_t)(2 * KVH * HD) * DIM);
  cvt(W_xproj, Wxp_h, (size_t)(DTR + 2 * NS) * II);
  cvt(W_dt, Wdt_h, (size_t)II * DTR);
  cvt(W_mo, Wmo_h, (size_t)DIM * II);
  cvt(W_proj, Wp_h, (size_t)DIM * DIM);

  // 2-3. input GEMMs
  gemm(xh, DIM, Wm_h, DIM, mp, MROWS, 2 * II, DIM);
  gemm(xh, DIM, Wq_h, DIM, q_lin, MROWS, DIM, DIM);
  gemm(xh, DIM, Wkv_h, DIM, kv_ln, MROWS, 2 * KVH * HD, DIM);

  // 4. RMSNorm + RoPE + V transpose
  qkv_prep<<<(MROWS * (NH + KVH)) / 8, 256, 0, stream>>>(q_lin, kv_ln, q_gain, qh, kh);
  v_prep<<<(BB * KVH * HD * SEQ) / 256, 256, 0, stream>>>(kv_ln, vt);

  // 5. flash attention
  flash_attn<<<(BB * NH * (SEQ / 16)) / 8, 256, 0, stream>>>(qh, kh, vt, attn);

  // 6. conv + silu
  conv_silu<<<dim3(MROWS, II / 256), 256, 0, stream>>>(mp, conv_w, conv_b, u, uh);

  // 7. x-proj GEMM + f16 copy of params
  gemm(uh, II, Wxp_h, II, prms, MROWS, DTR + 2 * NS, II);
  cvt(prms, prmsh, (size_t)MROWS * (DTR + 2 * NS));

  // 8-9. dt GEMM + softplus
  gemm(prmsh, DTR + 2 * NS, Wdt_h, DTR, delta, MROWS, II, DTR);
  softplus_bias<<<(MROWS * II) / 256, 256, 0, stream>>>(delta, b_dt, (size_t)MROWS * II);

  // 10. selective scan (async-to-LDS staging of B/C)
  ssm_scan<<<dim3(II / 256, BB), 256, 0, stream>>>(delta, prms, u, A_log, Dvec, mp, yh);

  // 11. mamba out GEMM
  gemm(yh, II, Wmo_h, II, mamba, MROWS, DIM, II);

  // 12. merge
  merge_kernel<<<(MROWS * DIM) / 256, 256, 0, stream>>>(attn, mamba, m_alpha, mrgd,
                                                        (size_t)MROWS * DIM);

  // 13. final projection -> d_out (f32)
  gemm(mrgd, DIM, Wp_h, DIM, (float*)d_out, MROWS, DIM, DIM);

  (void)in_sizes; (void)n_in; (void)out_size; (void)ws_size;
}
